// ClassifyTransformer_54546084659819
// MI455X (gfx1250) — matrix operand order repaired and checked by reading, not compiled, not run on hardware
//
#include <hip/hip_runtime.h>
#include <math.h>

typedef unsigned short u16;
typedef unsigned int   u32;
typedef __attribute__((ext_vector_type(4)))  unsigned int u32x4;
typedef __attribute__((ext_vector_type(4)))  float        f32x4;
typedef __attribute__((ext_vector_type(16))) __bf16       v16bf;
typedef __attribute__((ext_vector_type(8)))  float        v8f;

union Frag { u32x4 u[2]; v16bf v; };

static constexpr int Bb = 32, Ss = 512, Dd = 512, Hh = 8, FFf = 2048, Cc = 16;
static constexpr int NLAYER = 6;

// attention dynamic LDS layout
static constexpr int SC_BYTES = 16 * 520 * 4;   // fp32 scores, stride 520
static constexpr int PS_BYTES = 16 * 520 * 2;   // bf16 P
static constexpr int KS_BYTES = 128 * 72 * 2;   // K chunk, stride 72 elems (144B rows)
static constexpr int QS_BYTES = 16 * 72 * 2;    // Q tile
static constexpr int VT_BYTES = 64 * 40 * 2;    // V^T chunk, stride 40 elems (80B rows)
static constexpr int ATTN_SMEM = SC_BYTES + PS_BYTES + KS_BYTES + QS_BYTES + VT_BYTES;

__device__ __forceinline__ u16 f2bf(float f) {
  u32 u = __float_as_uint(f);
  u32 r = u + 0x7FFFu + ((u >> 16) & 1u);   // round-to-nearest-even
  return (u16)(r >> 16);
}

__device__ __forceinline__ v8f wmma_bf16(v16bf a, v16bf b, v8f c) {
  return __builtin_amdgcn_wmma_f32_16x16x32_bf16(false, a, false, b, (short)0, c,
                                                 false, false);
}

// gfx1250 async global->LDS copy (16B per lane), tracked by ASYNCcnt.
// LDS aperture truncates flat addr to [31:0] => low 32 bits are the LDS offset.
__device__ __forceinline__ void async_b128(const void* g, void* l) {
  u32 loff = (u32)(size_t)l;
  asm volatile("global_load_async_to_lds_b128 %0, %1, off"
               :: "v"(loff), "v"(g) : "memory");
}
__device__ __forceinline__ void wait_async() {
  asm volatile("s_wait_asynccnt 0x0" ::: "memory");
}

// ---------------------------------------------------------------------------
// Weight preprocessing: fp32 (L,K,N) -> bf16 transposed (L,N,K), once per call.
// 32x32 LDS tile transpose; coalesced reads and writes.
// ---------------------------------------------------------------------------
__global__ __launch_bounds__(256)
void transpose_cvt_kernel(const float* __restrict__ src, u16* __restrict__ dst,
                          int K, int N)
{
  __shared__ float tile[32][33];
  const int k0 = blockIdx.y * 32, n0 = blockIdx.x * 32;
  const float* s = src + (size_t)blockIdx.z * K * N;
  u16* d = dst + (size_t)blockIdx.z * K * N;
  const int tx = threadIdx.x & 31, ty = threadIdx.x >> 5;
  #pragma unroll
  for (int i = 0; i < 4; ++i) {
    int kk = k0 + ty + i * 8, nn = n0 + tx;
    tile[ty + i * 8][tx] = (kk < K && nn < N) ? s[(size_t)kk * N + nn] : 0.f;
  }
  __syncthreads();
  #pragma unroll
  for (int i = 0; i < 4; ++i) {
    int nn = n0 + ty + i * 8, kk = k0 + tx;
    if (nn < N && kk < K) d[(size_t)nn * K + kk] = f2bf(tile[tx][ty + i * 8]);
  }
}

// ---------------------------------------------------------------------------
// GEMM: out(M,N) = A_bf16(M,K) @ Wt_bf16(N,K)^T + bias (+resid) [+relu]
// 256 threads = 8 waves; block tile 128x128; wave tile 32x64; K step 32.
// Double-buffered LDS; BOTH tiles staged via async-to-LDS.
// ---------------------------------------------------------------------------
static constexpr int ASZ = 128 * 40;   // one buffer (u16 elems), 80B rows

__global__ __launch_bounds__(256)
void gemm_bf16_kernel(const u16* __restrict__ A, const u16* __restrict__ Wt,
                      const float* __restrict__ bias, const float* resid,
                      float* outF, u16* outB, int M, int N, int K, int relu)
{
  __shared__ __align__(16) u16 As[2 * ASZ];
  __shared__ __align__(16) u16 Ws[2 * ASZ];

  const int tid   = threadIdx.x;
  const int bn    = blockIdx.x, bm = blockIdx.y;
  const int wid   = tid >> 5, lane = tid & 31;
  const int wm    = wid >> 1, wn = wid & 1;
  const int lhalf = lane >> 4, lm = lane & 15;

  v8f acc[2][4];
  #pragma unroll
  for (int i = 0; i < 2; ++i)
    #pragma unroll
    for (int j = 0; j < 4; ++j)
      acc[i][j] = (v8f){0.f, 0.f, 0.f, 0.f, 0.f, 0.f, 0.f, 0.f};

  // staging geometry (identical for A rows and W^T rows): 128 rows x 32 cols
  const int rowS = tid >> 1, halfS = tid & 1;
  const int gm   = bm * 128 + rowS;         // A row
  const int gn   = bn * 128 + rowS;         // W^T row
  // pre-zero OOB rows once (async path skips masked lanes; zeros persist)
  {
    u32x4 z = {0, 0, 0, 0};
    #pragma unroll
    for (int bf = 0; bf < 2; ++bf) {
      if (gm >= M) {
        *(u32x4*)(&As[bf * ASZ + rowS * 40 + halfS * 16])     = z;
        *(u32x4*)(&As[bf * ASZ + rowS * 40 + halfS * 16 + 8]) = z;
      }
      if (gn >= N) {
        *(u32x4*)(&Ws[bf * ASZ + rowS * 40 + halfS * 16])     = z;
        *(u32x4*)(&Ws[bf * ASZ + rowS * 40 + halfS * 16 + 8]) = z;
      }
    }
  }

  auto stage = [&](int buf, int kc) {
    if (gm < M) {
      const u16* g = A + (size_t)gm * K + kc + halfS * 16;
      u16* l = &As[buf * ASZ + rowS * 40 + halfS * 16];
      async_b128(g, l);
      async_b128(g + 8, l + 8);
    }
    if (gn < N) {
      const u16* g = Wt + (size_t)gn * K + kc + halfS * 16;
      u16* l = &Ws[buf * ASZ + rowS * 40 + halfS * 16];
      async_b128(g, l);
      async_b128(g + 8, l + 8);
    }
  };

  const int nk = K >> 5;
  stage(0, 0);
  wait_async();
  __syncthreads();

  int cur = 0;
  for (int ki = 0; ki < nk; ++ki) {
    const bool has_next = (ki + 1) < nk;
    if (has_next) stage(cur ^ 1, (ki + 1) << 5);

    // ---- fragments + 8 WMMAs from buffer `cur` ----
    Frag a0, a1, b[4];
    {
      int r0 = cur * ASZ + (wm * 32 + lm) * 40 + lhalf * 8;
      a0.u[0] = *(const u32x4*)(&As[r0]);      a0.u[1] = *(const u32x4*)(&As[r0 + 16]);
      int r1 = cur * ASZ + (wm * 32 + 16 + lm) * 40 + lhalf * 8;
      a1.u[0] = *(const u32x4*)(&As[r1]);      a1.u[1] = *(const u32x4*)(&As[r1 + 16]);
      #pragma unroll
      for (int sn = 0; sn < 4; ++sn) {
        int c0 = cur * ASZ + (wn * 64 + sn * 16 + lm) * 40 + lhalf * 8;
        b[sn].u[0] = *(const u32x4*)(&Ws[c0]);
        b[sn].u[1] = *(const u32x4*)(&Ws[c0 + 16]);
      }
    }
    #pragma unroll
    for (int sn = 0; sn < 4; ++sn) {
      acc[0][sn] = wmma_bf16(a0.v, b[sn].v, acc[0][sn]);
      acc[1][sn] = wmma_bf16(a1.v, b[sn].v, acc[1][sn]);
    }

    if (has_next) wait_async();
    __syncthreads();
    cur ^= 1;
  }

  // ---- epilogue: bias (+relu) (+resid), fp32 and/or bf16 out ----
  const bool fullTile = ((bm * 128 + 128) <= M) && ((bn * 128 + 128) <= N);
  auto epi = [&](bool guard) {
    #pragma unroll
    for (int sm = 0; sm < 2; ++sm)
      #pragma unroll
      for (int sn = 0; sn < 4; ++sn)
        #pragma unroll
        for (int r = 0; r < 8; ++r) {
          int m = bm * 128 + wm * 32 + sm * 16 + lhalf * 8 + r;
          int n = bn * 128 + wn * 64 + sn * 16 + lm;
          if (!guard || (m < M && n < N)) {
            float vv = acc[sm][sn][r] + bias[n];
            if (relu) vv = fmaxf(vv, 0.0f);
            size_t off = (size_t)m * N + n;
            if (resid) vv += resid[off];
            if (outF) outF[off] = vv;
            if (outB) outB[off] = f2bf(vv);
          }
        }
  };
  if (fullTile) epi(false); else epi(true);
}

// ---------------------------------------------------------------------------
// Attention: one (b, h, 16-query-rows) tile per block; 128 threads = 4 waves.
// scores = QK^T/8 -> softmax -> P@V, all matmuls via WMMA bf16.
// Q and K tiles staged with async-to-LDS.
// ---------------------------------------------------------------------------
__global__ __launch_bounds__(128)
void attn_kernel(const u16* __restrict__ q, const u16* __restrict__ k,
                 const u16* __restrict__ v, u16* __restrict__ o, int Lq, int Lk)
{
  extern __shared__ __align__(16) char smem[];
  float* Sc = (float*)smem;
  u16*   Ps = (u16*)(smem + SC_BYTES);
  u16*   Ks = (u16*)(smem + SC_BYTES + PS_BYTES);
  u16*   Qs = (u16*)(smem + SC_BYTES + PS_BYTES + KS_BYTES);
  u16*   Vt = (u16*)(smem + SC_BYTES + PS_BYTES + KS_BYTES + QS_BYTES);

  const int tid = threadIdx.x;
  const int b = blockIdx.x / Hh, h = blockIdx.x % Hh;
  const int qbase = blockIdx.y * 16;
  const int wid = tid >> 5, lane = tid & 31;
  const int lhalf = lane >> 4, lm = lane & 15;
  const int nch = (Lk + 127) >> 7;
  const int Lc = nch << 7;

  // stage Q tile (16 x 64 bf16) asynchronously
  {
    int r = tid >> 3, d0 = (tid & 7) * 8;
    async_b128(q + ((size_t)(b * Lq + qbase + r)) * Dd + h * 64 + d0,
               &Qs[r * 72 + d0]);
  }

  // ---- scores: chunks of 128 keys ----
  for (int c = 0; c < nch; ++c) {
    __syncthreads();
    {
      int jg = c * 128 + tid;
      if (jg < Lk) {
        const u16* src = k + ((size_t)(b * Lk + jg)) * Dd + h * 64;
        #pragma unroll
        for (int i = 0; i < 8; ++i) async_b128(src + i * 8, &Ks[tid * 72 + i * 8]);
      } else {
        u32x4 z = {0, 0, 0, 0};
        #pragma unroll
        for (int i = 0; i < 8; ++i) *(u32x4*)(&Ks[tid * 72 + i * 8]) = z;
      }
    }
    wait_async();
    __syncthreads();

    Frag aq0, aq1;
    aq0.u[0] = *(const u32x4*)(&Qs[lm * 72 + lhalf * 8]);
    aq0.u[1] = *(const u32x4*)(&Qs[lm * 72 + lhalf * 8 + 16]);
    aq1.u[0] = *(const u32x4*)(&Qs[lm * 72 + 32 + lhalf * 8]);
    aq1.u[1] = *(const u32x4*)(&Qs[lm * 72 + 32 + lhalf * 8 + 16]);

    #pragma unroll
    for (int sn = 0; sn < 2; ++sn) {
      v8f acc = (v8f){0.f, 0.f, 0.f, 0.f, 0.f, 0.f, 0.f, 0.f};
      int krow = (wid * 32 + sn * 16 + lm) * 72;
      Frag bk0, bk1;
      bk0.u[0] = *(const u32x4*)(&Ks[krow + lhalf * 8]);
      bk0.u[1] = *(const u32x4*)(&Ks[krow + lhalf * 8 + 16]);
      bk1.u[0] = *(const u32x4*)(&Ks[krow + 32 + lhalf * 8]);
      bk1.u[1] = *(const u32x4*)(&Ks[krow + 32 + lhalf * 8 + 16]);
      acc = wmma_bf16(aq0.v, bk0.v, acc);
      acc = wmma_bf16(aq1.v, bk1.v, acc);
      #pragma unroll
      for (int r = 0; r < 8; ++r) {
        int m = lhalf * 8 + r;
        int j = c * 128 + wid * 32 + sn * 16 + lm;
        Sc[m * 520 + j] = (j < Lk) ? acc[r] * 0.125f : -1e30f;
      }
    }
  }
  __syncthreads();

  // ---- softmax: wave handles 4 rows ----
  for (int rr = 0; rr < 4; ++rr) {
    int m = wid * 4 + rr;
    float mx = -3e38f;
    for (int j = lane; j < Lc; j += 32) mx = fmaxf(mx, Sc[m * 520 + j]);
    #pragma unroll
    for (int s = 16; s >= 1; s >>= 1) mx = fmaxf(mx, __shfl_xor(mx, s, 32));
    float sum = 0.f;
    for (int j = lane; j < Lc; j += 32) {
      float e = __expf(Sc[m * 520 + j] - mx);
      Sc[m * 520 + j] = e;
      sum += e;
    }
    #pragma unroll
    for (int s = 16; s >= 1; s >>= 1) sum += __shfl_xor(sum, s, 32);
    float inv = 1.0f / sum;
    for (int j = lane; j < Lc; j += 32) Ps[m * 520 + j] = f2bf(Sc[m * 520 + j] * inv);
  }

  // ---- P @ V: chunks of 32 keys; wave owns 16 of the 64 head dims ----
  v8f accO = (v8f){0.f, 0.f, 0.f, 0.f, 0.f, 0.f, 0.f, 0.f};
  for (int c2 = 0; c2 < (Lc >> 5); ++c2) {
    __syncthreads();
    {
      int srow = tid >> 2, d0 = (tid & 3) * 16;
      int sg = c2 * 32 + srow;
      u16 tmp[16];
      if (sg < Lk) {
        const u32x4* src = (const u32x4*)(v + ((size_t)(b * Lk + sg)) * Dd + h * 64 + d0);
        *(u32x4*)&tmp[0] = src[0];
        *(u32x4*)&tmp[8] = src[1];
      } else {
        #pragma unroll
        for (int j = 0; j < 16; ++j) tmp[j] = 0;
      }
      #pragma unroll
      for (int j = 0; j < 16; ++j) Vt[(d0 + j) * 40 + srow] = tmp[j];
    }
    __syncthreads();
    Frag ap, bv;
    ap.u[0] = *(const u32x4*)(&Ps[lm * 520 + c2 * 32 + lhalf * 8]);
    ap.u[1] = *(const u32x4*)(&Ps[lm * 520 + c2 * 32 + lhalf * 8 + 16]);
    bv.u[0] = *(const u32x4*)(&Vt[(wid * 16 + lm) * 40 + lhalf * 8]);
    bv.u[1] = *(const u32x4*)(&Vt[(wid * 16 + lm) * 40 + lhalf * 8 + 16]);
    accO = wmma_bf16(ap.v, bv.v, accO);
  }
  #pragma unroll
  for (int r = 0; r < 8; ++r) {
    int m = lhalf * 8 + r;
    o[((size_t)(b * Lq + qbase + m)) * Dd + h * 64 + wid * 16 + lm] = f2bf(accO[r]);
  }
}

// ---------------------------------------------------------------------------
// LayerNorm (torch-style: (x-mean)/(std_unbiased+eps)), bf16 out; wave per row
// cols fixed = 512; lane owns 16 contiguous columns.
// ---------------------------------------------------------------------------
__global__ __launch_bounds__(256)
void ln_kernel(const float* __restrict__ x, u16* __restrict__ outB, int rows)
{
  int wid = threadIdx.x >> 5, lane = threadIdx.x & 31;
  int row = blockIdx.x * 8 + wid;
  if (row >= rows) return;
  const float* xr = x + (size_t)row * 512;
  const f32x4* xv = (const f32x4*)(xr + lane * 16);
  f32x4 vbuf[4];
  float s = 0.f, sq = 0.f;
  #pragma unroll
  for (int i = 0; i < 4; ++i) {
    vbuf[i] = xv[i];
    #pragma unroll
    for (int e = 0; e < 4; ++e) { float vv = vbuf[i][e]; s += vv; sq += vv * vv; }
  }
  #pragma unroll
  for (int i = 16; i >= 1; i >>= 1) { s += __shfl_xor(s, i, 32); sq += __shfl_xor(sq, i, 32); }
  float mean = s / 512.0f;
  float var  = fmaxf((sq - 512.0f * mean * mean) / 511.0f, 0.0f);
  float inv  = 1.0f / (sqrtf(var) + 1e-6f);
  u32* orow = (u32*)(outB + (size_t)row * 512) + lane * 8;
  #pragma unroll
  for (int j = 0; j < 8; ++j) {
    float e0 = vbuf[j >> 1][(2 * j) & 3];
    float e1 = vbuf[j >> 1][(2 * j + 1) & 3];
    u32 pk = (u32)f2bf((e0 - mean) * inv) | ((u32)f2bf((e1 - mean) * inv) << 16);
    orow[j] = pk;
  }
}

// ---------------------------------------------------------------------------
// Embeddings + sinusoidal positional encoding
// ---------------------------------------------------------------------------
__global__ void embed_enc_kernel(const float* __restrict__ emb, const int* __restrict__ scr,
                                 float* __restrict__ x, int total)
{
  int i = blockIdx.x * 256 + threadIdx.x;
  if (i >= total) return;
  int d = i & 511;
  int bs = i >> 9;
  int s = bs & 511;
  int tok = scr[bs];
  int e = d & ~1;
  float freq = __expf(-(float)e * (9.21034037f / 512.0f));
  float ang = (float)s * freq;
  float pe = (d & 1) ? __cosf(ang) : __sinf(ang);
  x[i] = emb[(size_t)tok * 512 + d] * 22.627417f + pe;
}

__global__ void embed_dec_kernel(const float* __restrict__ emb, float* __restrict__ y, int total)
{
  int i = blockIdx.x * 256 + threadIdx.x;
  if (i >= total) return;
  int d = i & 511;
  int c = (i >> 9) & 15;
  int e = d & ~1;
  float freq = __expf(-(float)e * (9.21034037f / 512.0f));
  float ang = (float)c * freq;
  float pe = (d & 1) ? __cosf(ang) : __sinf(ang);
  y[i] = emb[(size_t)c * 512 + d] * 22.627417f + pe;
}

__global__ void logsoftmax_kernel(const float* __restrict__ lg, float* __restrict__ out)
{
  int row = threadIdx.x;
  if (row >= 32) return;
  const float* r = lg + row * 16;
  float m = r[0];
  for (int j = 1; j < 16; ++j) m = fmaxf(m, r[j]);
  float s = 0.f;
  for (int j = 0; j < 16; ++j) s += __expf(r[j] - m);
  float lse = m + logf(s);
  for (int j = 0; j < 16; ++j) out[row * 16 + j] = r[j] - lse;
}

// ---------------------------------------------------------------------------
// Host orchestration
// ---------------------------------------------------------------------------
static inline void gemmL(hipStream_t st, const u16* A, const u16* Wt, const float* bias,
                         const float* resid, float* outF, u16* outB,
                         int M, int N, int K, int relu)
{
  dim3 g((N + 127) / 128, (M + 127) / 128);
  gemm_bf16_kernel<<<g, 256, 0, st>>>(A, Wt, bias, resid, outF, outB, M, N, K, relu);
}

static inline void attnL(hipStream_t st, const u16* q, const u16* k, const u16* v,
                         u16* o, int Lq, int Lk)
{
  dim3 g(Bb * Hh, Lq / 16);
  attn_kernel<<<g, 128, ATTN_SMEM, st>>>(q, k, v, o, Lq, Lk);
}

extern "C" void kernel_launch(void* const* d_in, const int* in_sizes, int n_in,
                              void* d_out, int out_size, void* d_ws, size_t ws_size,
                              hipStream_t stream)
{
  const float* src_emb = (const float*)d_in[0];
  const float* tgt_emb = (const float*)d_in[1];
  auto Wp = [&](int g, int j) { return (const float*)d_in[2 + g * 8 + j * 2]; };
  auto Bp = [&](int g, int j) { return (const float*)d_in[3 + g * 8 + j * 2]; };
  const float* e_w1 = (const float*)d_in[26];
  const float* e_b1 = (const float*)d_in[27];
  const float* e_w2 = (const float*)d_in[28];
  const float* e_b2 = (const float*)d_in[29];
  const float* dw1  = (const float*)d_in[30];
  const float* db1  = (const float*)d_in[31];
  const float* dw2  = (const float*)d_in[32];
  const float* db2  = (const float*)d_in[33];
  const float* gen_w = (const float*)d_in[34];
  const float* gen_b = (const float*)d_in[35];
  const int*   scr   = (const int*)d_in[36];
  float* out = (float*)d_out;

  char* wsp = (char*)d_ws;
  auto carve = [&](size_t bytes) -> char* {
    char* p = wsp;
    wsp += (bytes + 255) & ~(size_t)255;
    return p;
  };
  const int Mtok = Bb * Ss;   // 16384
  const int Mdec = Bb * Cc;   // 512
  float* x    = (float*)carve((size_t)Mtok * Dd * 4);
  u16*   t    = (u16*)carve((size_t)Mtok * Dd * 2);
  u16*   qb   = (u16*)carve((size_t)Mtok * Dd * 2);
  u16*   kb   = (u16*)carve((size_t)Mtok * Dd * 2);
  u16*   vb   = (u16*)carve((size_t)Mtok * Dd * 2);
  u16*   ob   = (u16*)carve((size_t)Mtok * Dd * 2);
  u16*   hf   = (u16*)carve((size_t)Mtok * FFf * 2);
  u16*   encB = (u16*)carve((size_t)Mtok * Dd * 2);
  float* y    = (float*)carve((size_t)Mdec * Dd * 4);
  u16*   td   = (u16*)carve((size_t)Mdec * Dd * 2);
  u16*   decB = (u16*)carve((size_t)Mdec * Dd * 2);
  float* logits = (float*)carve((size_t)Bb * Cc * 4);

  // ---- pre-convert all weights: fp32 (L,K,N) -> bf16 transposed (L,N,K) ----
  auto cvtW = [&](const float* src, int L, int K, int N) -> u16* {
    u16* dst = (u16*)carve((size_t)L * K * N * 2);
    dim3 g((N + 31) / 32, (K + 31) / 32, L);
    transpose_cvt_kernel<<<g, 256, 0, stream>>>(src, dst, K, N);
    return dst;
  };
  u16* Wt[3][4];                       // [e/d/s][q/k/v/o] transposed bf16
  for (int g = 0; g < 3; ++g)
    for (int j = 0; j < 4; ++j)
      Wt[g][j] = cvtW(Wp(g, j), NLAYER, Dd, Dd);
  u16* eW1t = cvtW(e_w1, NLAYER, Dd, FFf);
  u16* eW2t = cvtW(e_w2, NLAYER, FFf, Dd);
  u16* dW1t = cvtW(dw1,  NLAYER, Dd, FFf);
  u16* dW2t = cvtW(dw2,  NLAYER, FFf, Dd);
  u16* genWt = cvtW(gen_w, 1, Cc * Dd, Cc);
  const size_t aStride = (size_t)Dd * Dd;       // per-layer elems (attn)
  const size_t fStride = (size_t)Dd * FFf;      // per-layer elems (ffn)

  // ---------------- encoder ----------------
  embed_enc_kernel<<<(Mtok * Dd + 255) / 256, 256, 0, stream>>>(src_emb, scr, x, Mtok * Dd);
  for (int i = 0; i < NLAYER; ++i) {
    size_t wo = (size_t)i * aStride, bo = (size_t)i * Dd;
    ln_kernel<<<Mtok / 8, 256, 0, stream>>>(x, t, Mtok);
    gemmL(stream, t, Wt[0][0] + wo, Bp(0, 0) + bo, nullptr, nullptr, qb, Mtok, Dd, Dd, 0);
    gemmL(stream, t, Wt[0][1] + wo, Bp(0, 1) + bo, nullptr, nullptr, kb, Mtok, Dd, Dd, 0);
    gemmL(stream, t, Wt[0][2] + wo, Bp(0, 2) + bo, nullptr, nullptr, vb, Mtok, Dd, Dd, 0);
    attnL(stream, qb, kb, vb, ob, Ss, Ss);
    gemmL(stream, ob, Wt[0][3] + wo, Bp(0, 3) + bo, x, x, nullptr, Mtok, Dd, Dd, 0);
    ln_kernel<<<Mtok / 8, 256, 0, stream>>>(x, t, Mtok);
    gemmL(stream, t, eW1t + (size_t)i * fStride, e_b1 + (size_t)i * FFf,
          nullptr, nullptr, hf, Mtok, FFf, Dd, 1);
    gemmL(stream, hf, eW2t + (size_t)i * fStride, e_b2 + (size_t)i * Dd,
          x, x, nullptr, Mtok, Dd, FFf, 0);
  }
  ln_kernel<<<Mtok / 8, 256, 0, stream>>>(x, encB, Mtok);

  // ---------------- decoder ----------------
  embed_dec_kernel<<<(Mdec * Dd + 255) / 256, 256, 0, stream>>>(tgt_emb, y, Mdec * Dd);
  for (int i = 0; i < NLAYER; ++i) {
    size_t wo = (size_t)i * aStride, bo = (size_t)i * Dd;
    // self attention (Lq = Lk = 16)
    ln_kernel<<<Mdec / 8, 256, 0, stream>>>(y, td, Mdec);
    gemmL(stream, td, Wt[1][0] + wo, Bp(1, 0) + bo, nullptr, nullptr, qb, Mdec, Dd, Dd, 0);
    gemmL(stream, td, Wt[1][1] + wo, Bp(1, 1) + bo, nullptr, nullptr, kb, Mdec, Dd, Dd, 0);
    gemmL(stream, td, Wt[1][2] + wo, Bp(1, 2) + bo, nullptr, nullptr, vb, Mdec, Dd, Dd, 0);
    attnL(stream, qb, kb, vb, ob, Cc, Cc);
    gemmL(stream, ob, Wt[1][3] + wo, Bp(1, 3) + bo, y, y, nullptr, Mdec, Dd, Dd, 0);
    // cross attention (Lq = 16, Lk = S)
    ln_kernel<<<Mdec / 8, 256, 0, stream>>>(y, td, Mdec);
    gemmL(stream, td,   Wt[2][0] + wo, Bp(2, 0) + bo, nullptr, nullptr, qb, Mdec, Dd, Dd, 0);
    gemmL(stream, encB, Wt[2][1] + wo, Bp(2, 1) + bo, nullptr, nullptr, kb, Mtok, Dd, Dd, 0);
    gemmL(stream, encB, Wt[2][2] + wo, Bp(2, 2) + bo, nullptr, nullptr, vb, Mtok, Dd, Dd, 0);
    attnL(stream, qb, kb, vb, ob, Cc, Ss);
    gemmL(stream, ob, Wt[2][3] + wo, Bp(2, 3) + bo, y, y, nullptr, Mdec, Dd, Dd, 0);
    // FFN
    ln_kernel<<<Mdec / 8, 256, 0, stream>>>(y, td, Mdec);
    gemmL(stream, td, dW1t + (size_t)i * fStride, db1 + (size_t)i * FFf,
          nullptr, nullptr, hf, Mdec, FFf, Dd, 1);
    gemmL(stream, hf, dW2t + (size_t)i * fStride, db2 + (size_t)i * Dd,
          y, y, nullptr, Mdec, Dd, FFf, 0);
  }
  ln_kernel<<<Mdec / 8, 256, 0, stream>>>(y, decB, Mdec);

  // generator: (32, 8192) @ (8192, 16) + bias, then log_softmax
  gemmL(stream, decB, genWt, gen_b, nullptr, logits, nullptr, Bb, Cc, Cc * Dd, 0);
  logsoftmax_kernel<<<1, 32, 0, stream>>>(logits, out);
}
